// SelfAttention_84756884620114
// MI455X (gfx1250) — compile-verified, matrix operand order checked by reading, not run on hardware
//
#include <hip/hip_runtime.h>
#include <hip/hip_bf16.h>
#include <cmath>
#include <cstdint>

typedef __bf16 bf16;
typedef __attribute__((ext_vector_type(16))) __bf16 v16bf;
typedef __attribute__((ext_vector_type(8)))  float  v8f;
typedef __attribute__((ext_vector_type(4)))  unsigned int u32x4;
typedef __attribute__((ext_vector_type(8)))  int i32x8;
typedef __attribute__((ext_vector_type(4)))  int i32x4;

#define DEV __device__ __forceinline__

constexpr int Lq   = 2048;
constexpr int DIMc = 3072;
constexpr int Hc   = 24;
constexpr int Dc   = 128;
constexpr int N3   = 3 * DIMc;   // 9216

DEV v8f zero8() { v8f z = {0.f,0.f,0.f,0.f,0.f,0.f,0.f,0.f}; return z; }

DEV v8f wmma_bf16(v16bf a, v16bf b, v8f c) {
  return __builtin_amdgcn_wmma_f32_16x16x32_bf16(false, a, false, b, (short)0, c,
                                                 false, false);
}

// Fragment load, 16x32 bf16, K contiguous per lane (works for A-layout rows
// and for B-layout columns when the tile is staged K-major per column).
// Caller passes base = linePtr + (lane>>4)*8; reads elements [0..7],[16..23].
DEV v16bf load_frag(const bf16* base) {
  v16bf v;
#pragma unroll
  for (int i = 0; i < 8; ++i) v[i] = base[i];
#pragma unroll
  for (int i = 0; i < 8; ++i) v[8 + i] = base[16 + i];
  return v;
}

// --------------------------------------------------------------- TDM issue
// Build a 2D Tensor DMA Descriptor (D#) and issue TENSOR_LOAD_TO_LDS.
// 128x128 bf16 tile; dim0 = contiguous axis. Groups per ISA 08 §8.3/8.4.
// This toolchain exposes the 6-arg builtin: (g0, g1, g2, g3, ext, cpol).
DEV void tdm_load_tile_2d(unsigned lds_off, const bf16* gptr,
                          unsigned tensor_d0, unsigned tensor_d1,
                          unsigned stride0_elems) {
  unsigned long long ga = (unsigned long long)(uintptr_t)gptr;
  u32x4 g0;
  g0[0] = 1u;                                   // count=1, user descriptor
  g0[1] = lds_off;                              // LDS byte offset
  g0[2] = (unsigned)ga;                         // global_addr[31:0]
  g0[3] = (unsigned)((ga >> 32) & 0x1ffffffu)   // global_addr[56:32]
          | (2u << 30);                         // type = 2 ("image")
  i32x8 g1;
  g1[0] = (int)(1u << 16);                      // data_size=1 (2 bytes)
  g1[1] = (int)((tensor_d0 & 0xffffu) << 16);   // tensor_dim0[15:0] @ bit48
  g1[2] = (int)((tensor_d0 >> 16) | ((tensor_d1 & 0xffffu) << 16));
  g1[3] = (int)((tensor_d1 >> 16) | (128u << 16));   // tile_dim0 = 128
  g1[4] = 128;                                  // tile_dim1=128, tile_dim2=0
  g1[5] = (int)stride0_elems;                   // tensor_dim0_stride[31:0]
  g1[6] = 0;
  g1[7] = 0;
  i32x4 gz4 = {0, 0, 0, 0};                     // groups 2/3: 2D tensor
  i32x8 gz8 = {0, 0, 0, 0, 0, 0, 0, 0};         // extension words (unused)
  __builtin_amdgcn_tensor_load_to_lds(g0, g1, gz4, gz4, gz8, 0);
}

// ---------------------------------------------------------------- converts
__global__ void cvt_f32_bf16(const float* __restrict__ in, bf16* __restrict__ out,
                             int n) {
  int i = blockIdx.x * blockDim.x + threadIdx.x;
  int stride = gridDim.x * blockDim.x;
  for (; i < n; i += stride) out[i] = (bf16)in[i];
}

// in: [K][N] f32 -> out: [N][K] bf16 (coalesced both sides via LDS tile)
__global__ __launch_bounds__(256) void cvt_t_f32_bf16(
    const float* __restrict__ in, bf16* __restrict__ out, int K, int N) {
  __shared__ float t[32][33];
  const int n0 = blockIdx.x * 32, k0 = blockIdx.y * 32;
  const int tx = threadIdx.x & 31, ty = threadIdx.x >> 5;
#pragma unroll
  for (int j = 0; j < 4; ++j)
    t[ty + j * 8][tx] = in[(size_t)(k0 + ty + j * 8) * N + n0 + tx];
  __syncthreads();
#pragma unroll
  for (int j = 0; j < 4; ++j)
    out[(size_t)(n0 + ty + j * 8) * K + k0 + tx] = (bf16)t[tx][ty + j * 8];
}

// --------------------------------------------------- QKV GEMM + RMSNorm + RoPE
// W pre-transposed: Wt[N3][DIMc]. Tile 128x128; 8 waves of 32x64 sub-tiles.
__global__ __launch_bounds__(256) void qkv_gemm_kernel(
    const bf16* __restrict__ X, const bf16* __restrict__ Wt,
    const float* __restrict__ qs, const float* __restrict__ ksc,
    const float* __restrict__ pe, bf16* __restrict__ Qo, bf16* __restrict__ Ko,
    bf16* __restrict__ Vt) {
  __shared__ bf16 As[128][48];   // 128 rows x 32 k, padded stride
  __shared__ bf16 Bt[128][48];   // 128 cols x 32 k (K-major per column)
  __shared__ float rowstat[128];

  const int rowBase = blockIdx.x * 128;
  const int colBase = blockIdx.y * 128;
  const int tid = threadIdx.x, lane = tid & 31, wave = tid >> 5;
  const int wy = wave >> 1, wx = wave & 1;
  const int laneM = lane & 15, laneH = lane >> 4;
  const int rofs = laneH * 8;

  v8f acc[2][4];
#pragma unroll
  for (int fr = 0; fr < 2; ++fr)
#pragma unroll
    for (int fc = 0; fc < 4; ++fc) acc[fr][fc] = zero8();

  const int ar = tid >> 1, as_ = (tid & 1) * 16;
  const bf16* aSrc = X + (size_t)(rowBase + ar) * DIMc + as_;
  const bf16* bSrc = Wt + (size_t)(colBase + ar) * DIMc + as_;

  for (int k0 = 0; k0 < DIMc; k0 += 32) {
    ((uint4*)&As[ar][as_])[0] = ((const uint4*)aSrc)[0];
    ((uint4*)&As[ar][as_])[1] = ((const uint4*)aSrc)[1];
    ((uint4*)&Bt[ar][as_])[0] = ((const uint4*)bSrc)[0];
    ((uint4*)&Bt[ar][as_])[1] = ((const uint4*)bSrc)[1];
    aSrc += 32;
    bSrc += 32;
    if (k0 + 32 < DIMc) {                      // gfx1250 global_prefetch_b8
      __builtin_prefetch(aSrc, 0, 0);
      __builtin_prefetch(bSrc, 0, 0);
    }
    __syncthreads();
    v16bf af[2], bw[4];
#pragma unroll
    for (int fr = 0; fr < 2; ++fr)
      af[fr] = load_frag(&As[wy * 32 + fr * 16 + laneM][laneH * 8]);
#pragma unroll
    for (int fc = 0; fc < 4; ++fc)
      bw[fc] = load_frag(&Bt[wx * 64 + fc * 16 + laneM][laneH * 8]);
#pragma unroll
    for (int fr = 0; fr < 2; ++fr)
#pragma unroll
      for (int fc = 0; fc < 4; ++fc)
        acc[fr][fc] = wmma_bf16(af[fr], bw[fc], acc[fr][fc]);
    __syncthreads();
  }

  const int sel = colBase / DIMc;           // 0=q, 1=k, 2=v
  const int head = (colBase % DIMc) / Dc;

  if (sel == 2) {
    // store V transposed: Vt[h][d][token] so PV B-fragments read contiguously
#pragma unroll
    for (int fr = 0; fr < 2; ++fr)
#pragma unroll
      for (int fc = 0; fc < 4; ++fc)
#pragma unroll
        for (int r = 0; r < 8; ++r) {
          int row = rowBase + wy * 32 + fr * 16 + rofs + r;
          int d = wx * 64 + fc * 16 + laneM;
          Vt[((size_t)head * Dc + d) * Lq + row] = (bf16)acc[fr][fc][r];
        }
  } else {
    const float* scv = (sel == 0) ? qs : ksc;
    bf16* dst = (sel == 0) ? Qo : Ko;
    if (tid < 128) rowstat[tid] = 0.f;
    __syncthreads();
#pragma unroll
    for (int fr = 0; fr < 2; ++fr)
#pragma unroll
      for (int fc = 0; fc < 4; ++fc)
#pragma unroll
        for (int r = 0; r < 8; ++r) {
          float v = acc[fr][fc][r];
          atomicAdd(&rowstat[wy * 32 + fr * 16 + rofs + r], v * v);  // ds_add_f32
        }
    __syncthreads();
#pragma unroll
    for (int fr = 0; fr < 2; ++fr)
#pragma unroll
      for (int fc = 0; fc < 4; ++fc)
#pragma unroll
        for (int r = 0; r < 8; ++r) {
          int lr = wy * 32 + fr * 16 + rofs + r;
          int row = rowBase + lr;
          int d = wx * 64 + fc * 16 + laneM;
          float rstd = rsqrtf(rowstat[lr] * (1.0f / 128.0f) + 1e-6f);
          float val = acc[fr][fc][r] * rstd * scv[d];
          float partner = __shfl_xor(val, 1, 32);   // RoPE pair (d^1), wave32
          const float* pep = pe + (size_t)row * 256 + (d >> 1) * 4 + (d & 1) * 2;
          float o = (d & 1) ? (pep[0] * partner + pep[1] * val)
                            : (pep[0] * val + pep[1] * partner);
          dst[((size_t)head * Lq + row) * Dc + d] = (bf16)o;
        }
  }
}

// ------------------------------------------------------------- flash attention
// One workgroup per (128-query block, head); 8 waves each own 16 query rows.
// K/V tiles staged to LDS by the Tensor Data Mover (wave 0 issues, TENSORcnt
// ordered: wait<=1 releases K early, wait 0 before PV).
__global__ __launch_bounds__(256) void attn_kernel(const bf16* __restrict__ Q,
                                                   const bf16* __restrict__ K,
                                                   const bf16* __restrict__ Vt,
                                                   bf16* __restrict__ O) {
  __shared__ bf16 Ks[128][128];      // [key][d]   32 KB (TDM)
  __shared__ bf16 Vs[128][128];      // [d][key]   32 KB (TDM, from Vt)
  __shared__ bf16 Ps[8][16 * 128];   // per-wave P staging, 32 KB

  const int h = blockIdx.y;
  const int qb = blockIdx.x;
  const int tid = threadIdx.x, lane = tid & 31, wave = tid >> 5;
  const int laneM = lane & 15, laneH = lane >> 4;
  const int rofs = laneH * 8;
  const int qRow0 = qb * 128 + wave * 16;
  const float scl = 0.08838834764831845f;   // 1/sqrt(128)

  const unsigned ks_off = (unsigned)(uintptr_t)&Ks[0][0];
  const unsigned vs_off = (unsigned)(uintptr_t)&Vs[0][0];

  v16bf qf[4];
  {
    const bf16* qp = Q + ((size_t)h * Lq + qRow0 + laneM) * Dc + laneH * 8;
#pragma unroll
    for (int c = 0; c < 4; ++c) qf[c] = load_frag(qp + c * 32);
  }

  float mrow[8], lrow[8];
  v8f of[8];
#pragma unroll
  for (int r = 0; r < 8; ++r) { mrow[r] = -INFINITY; lrow[r] = 0.f; }
#pragma unroll
  for (int nb = 0; nb < 8; ++nb) of[nb] = zero8();

  for (int kb = 0; kb < Lq / 128; ++kb) {
    const int kBase = kb * 128;
    __syncthreads();   // Ks/Vs reuse guard
    if (wave == 0) {
      // K tile: rows=keys, contiguous d (dim0 = d)
      tdm_load_tile_2d(ks_off, K + ((size_t)h * Lq + kBase) * Dc, Dc, Lq, Dc);
      // V tile from Vt: rows=d, contiguous keys (dim0 = key)
      tdm_load_tile_2d(vs_off, Vt + (size_t)h * Dc * Lq + kBase, Lq, Dc, Lq);
      __builtin_amdgcn_s_wait_tensorcnt(1);   // in-order: K tile complete
    }
    __syncthreads();   // Ks visible to all waves

    // S = Q * K^T : B-frag lane n -> Ks row (key), elements = d (contiguous)
    v8f sf[8];
#pragma unroll
    for (int nb = 0; nb < 8; ++nb) {
      v8f c = zero8();
      const bf16* kp = &Ks[nb * 16 + laneM][laneH * 8];
#pragma unroll
      for (int kc = 0; kc < 4; ++kc)
        c = wmma_bf16(qf[kc], load_frag(kp + kc * 32), c);
      sf[nb] = c;
    }

    // online softmax (rows live in one 16-lane half; reduce with shfl_xor)
    float rmax[8];
#pragma unroll
    for (int r = 0; r < 8; ++r) rmax[r] = -INFINITY;
#pragma unroll
    for (int nb = 0; nb < 8; ++nb)
#pragma unroll
      for (int r = 0; r < 8; ++r) {
        sf[nb][r] *= scl;
        rmax[r] = fmaxf(rmax[r], sf[nb][r]);
      }
#pragma unroll
    for (int r = 0; r < 8; ++r)
#pragma unroll
      for (int msk = 1; msk < 16; msk <<= 1)
        rmax[r] = fmaxf(rmax[r], __shfl_xor(rmax[r], msk, 32));

    float alpha[8], rsum[8];
#pragma unroll
    for (int r = 0; r < 8; ++r) {
      float mnew = fmaxf(mrow[r], rmax[r]);
      alpha[r] = __expf(mrow[r] - mnew);
      mrow[r] = mnew;
      rsum[r] = 0.f;
    }
#pragma unroll
    for (int nb = 0; nb < 8; ++nb)
#pragma unroll
      for (int r = 0; r < 8; ++r) {
        float p = __expf(sf[nb][r] - mrow[r]);
        sf[nb][r] = p;
        rsum[r] += p;
      }
#pragma unroll
    for (int r = 0; r < 8; ++r) {
#pragma unroll
      for (int msk = 1; msk < 16; msk <<= 1)
        rsum[r] += __shfl_xor(rsum[r], msk, 32);
      lrow[r] = lrow[r] * alpha[r] + rsum[r];
    }
#pragma unroll
    for (int nb = 0; nb < 8; ++nb)
#pragma unroll
      for (int r = 0; r < 8; ++r) of[nb][r] *= alpha[r];

    // restage P: C-layout -> row-major [16 q-rows][128 keys] (private region)
    bf16* Pw = &Ps[wave][0];
#pragma unroll
    for (int nb = 0; nb < 8; ++nb)
#pragma unroll
      for (int r = 0; r < 8; ++r)
        Pw[(rofs + r) * 128 + nb * 16 + laneM] = (bf16)sf[nb][r];

    if (wave == 0) __builtin_amdgcn_s_wait_tensorcnt(0);  // V tile complete
    __syncthreads();   // Vs visible to all waves

    // O += P * V : B-frag lane n -> Vs row (d), elements = keys (contiguous)
    const bf16* Pr = &Ps[wave][0] + laneM * 128 + laneH * 8;
#pragma unroll
    for (int kc = 0; kc < 4; ++kc) {
      v16bf pa = load_frag(Pr + kc * 32);
#pragma unroll
      for (int nb = 0; nb < 8; ++nb) {
        v16bf vb = load_frag(&Vs[nb * 16 + laneM][kc * 32 + laneH * 8]);
        of[nb] = wmma_bf16(pa, vb, of[nb]);
      }
    }
  }

#pragma unroll
  for (int r = 0; r < 8; ++r) {
    float inv = 1.f / lrow[r];
    int tok = qRow0 + rofs + r;
#pragma unroll
    for (int nb = 0; nb < 8; ++nb) {
      int d = nb * 16 + laneM;
      O[(size_t)tok * DIMc + h * Dc + d] = (bf16)(of[nb][r] * inv);
    }
  }
}

// ------------------------------------------------------------- output proj
// W pre-transposed: Wt[DIMc][DIMc].
__global__ __launch_bounds__(256) void proj_gemm_kernel(
    const bf16* __restrict__ A, const bf16* __restrict__ Wt,
    const float* __restrict__ bias, float* __restrict__ out) {
  __shared__ bf16 As[128][48];
  __shared__ bf16 Bt[128][48];

  const int rowBase = blockIdx.x * 128;
  const int colBase = blockIdx.y * 128;
  const int tid = threadIdx.x, lane = tid & 31, wave = tid >> 5;
  const int wy = wave >> 1, wx = wave & 1;
  const int laneM = lane & 15, laneH = lane >> 4;
  const int rofs = laneH * 8;

  v8f acc[2][4];
#pragma unroll
  for (int fr = 0; fr < 2; ++fr)
#pragma unroll
    for (int fc = 0; fc < 4; ++fc) acc[fr][fc] = zero8();

  const int ar = tid >> 1, as_ = (tid & 1) * 16;
  const bf16* aSrc = A + (size_t)(rowBase + ar) * DIMc + as_;
  const bf16* bSrc = Wt + (size_t)(colBase + ar) * DIMc + as_;

  for (int k0 = 0; k0 < DIMc; k0 += 32) {
    ((uint4*)&As[ar][as_])[0] = ((const uint4*)aSrc)[0];
    ((uint4*)&As[ar][as_])[1] = ((const uint4*)aSrc)[1];
    ((uint4*)&Bt[ar][as_])[0] = ((const uint4*)bSrc)[0];
    ((uint4*)&Bt[ar][as_])[1] = ((const uint4*)bSrc)[1];
    aSrc += 32;
    bSrc += 32;
    if (k0 + 32 < DIMc) {
      __builtin_prefetch(aSrc, 0, 0);
      __builtin_prefetch(bSrc, 0, 0);
    }
    __syncthreads();
    v16bf af[2], bw[4];
#pragma unroll
    for (int fr = 0; fr < 2; ++fr)
      af[fr] = load_frag(&As[wy * 32 + fr * 16 + laneM][laneH * 8]);
#pragma unroll
    for (int fc = 0; fc < 4; ++fc)
      bw[fc] = load_frag(&Bt[wx * 64 + fc * 16 + laneM][laneH * 8]);
#pragma unroll
    for (int fr = 0; fr < 2; ++fr)
#pragma unroll
      for (int fc = 0; fc < 4; ++fc)
        acc[fr][fc] = wmma_bf16(af[fr], bw[fc], acc[fr][fc]);
    __syncthreads();
  }

#pragma unroll
  for (int fr = 0; fr < 2; ++fr)
#pragma unroll
    for (int fc = 0; fc < 4; ++fc)
#pragma unroll
      for (int r = 0; r < 8; ++r) {
        int row = rowBase + wy * 32 + fr * 16 + rofs + r;
        int col = colBase + wx * 64 + fc * 16 + laneM;
        out[(size_t)row * DIMc + col] = acc[fr][fc][r] + bias[col];
      }
}

// ------------------------------------------------------------------ launch
extern "C" void kernel_launch(void* const* d_in, const int* in_sizes, int n_in,
                              void* d_out, int out_size, void* d_ws,
                              size_t ws_size, hipStream_t stream) {
  (void)in_sizes; (void)n_in; (void)out_size; (void)ws_size;
  const float* x      = (const float*)d_in[0];
  const float* pe     = (const float*)d_in[1];
  const float* w_qkv  = (const float*)d_in[2];
  const float* qscale = (const float*)d_in[3];
  const float* kscale = (const float*)d_in[4];
  const float* w_proj = (const float*)d_in[5];
  const float* b_proj = (const float*)d_in[6];
  float* out = (float*)d_out;

  char* ws = (char*)d_ws;
  size_t off = 0;
  bf16* xb     = (bf16*)(ws + off); off += (size_t)Lq * DIMc * 2;    // 12 MB
  bf16* wqkvt  = (bf16*)(ws + off); off += (size_t)DIMc * N3 * 2;    // 57 MB  [N3][DIMc]
  bf16* wprojt = (bf16*)(ws + off); off += (size_t)DIMc * DIMc * 2;  // 19 MB  [DIMc][DIMc]
  bf16* qb     = (bf16*)(ws + off); off += (size_t)Hc * Lq * Dc * 2; // [h][l][d]
  bf16* kb     = (bf16*)(ws + off); off += (size_t)Hc * Lq * Dc * 2; // [h][l][d]
  bf16* vt     = (bf16*)(ws + off); off += (size_t)Hc * Lq * Dc * 2; // [h][d][l]
  bf16* attnb  = (bf16*)(ws + off); off += (size_t)Lq * DIMc * 2;

  cvt_f32_bf16<<<1024, 256, 0, stream>>>(x, xb, Lq * DIMc);
  cvt_t_f32_bf16<<<dim3(N3 / 32, DIMc / 32), 256, 0, stream>>>(w_qkv, wqkvt,
                                                               DIMc, N3);
  cvt_t_f32_bf16<<<dim3(DIMc / 32, DIMc / 32), 256, 0, stream>>>(w_proj, wprojt,
                                                                 DIMc, DIMc);

  qkv_gemm_kernel<<<dim3(Lq / 128, N3 / 128), 256, 0, stream>>>(
      xb, wqkvt, qscale, kscale, pe, qb, kb, vt);

  attn_kernel<<<dim3(Lq / 128, Hc), 256, 0, stream>>>(qb, kb, vt, attnb);

  proj_gemm_kernel<<<dim3(Lq / 128, DIMc / 128), 256, 0, stream>>>(
      attnb, wprojt, b_proj, out);
}
